// UpSample_77841987273316
// MI455X (gfx1250) — matrix operand order repaired and checked by reading, not compiled, not run on hardware
//
#include <hip/hip_runtime.h>
#include <math.h>

// ---------------------------------------------------------------------------
// Shapes (fixed by the reference): B=8, N=1024, K=17, RATIO=4
// ---------------------------------------------------------------------------
#define BATCH   8
#define NPTS    1024
#define KNN_K   17
#define T_TOK   8192          // B*N
#define T4_TOK  32768         // B*N*4

// Async global->LDS (gfx1250): use if the toolchain declares the builtins.
#if defined(__has_builtin)
#  if __has_builtin(__builtin_amdgcn_global_load_async_to_lds_b128) && \
      __has_builtin(__builtin_amdgcn_s_wait_asynccnt)
#    define HAVE_ASYNC_LDS 1
#  endif
#endif
#ifndef HAVE_ASYNC_LDS
#  define HAVE_ASYNC_LDS 0
#endif

typedef __attribute__((ext_vector_type(16))) __bf16 v16bf;
typedef __attribute__((ext_vector_type(8)))  float  v8f;
typedef int v4i_async __attribute__((vector_size(16)));
typedef __attribute__((address_space(1))) v4i_async* gptr_v4i;
typedef __attribute__((address_space(3))) v4i_async* lptr_v4i;

static __device__ __forceinline__ __bf16 f2bf(float f) {
  unsigned int u = __builtin_bit_cast(unsigned int, f);
  unsigned int r = (u + 0x7FFFu + ((u >> 16) & 1u)) >> 16;
  unsigned short s = (unsigned short)r;
  return __builtin_bit_cast(__bf16, s);
}
static __device__ __forceinline__ float gelu_f(float x) {
  return 0.5f * x * (1.0f + erff(x * 0.70710678118654752440f));
}

// 16 floats from two 8-float runs -> bf16 fragment (4x global_load_b128)
static __device__ __forceinline__ v16bf cvt16(const float* __restrict__ p,
                                              const float* __restrict__ q) {
  const float4 x0 = *(const float4*)(p);
  const float4 x1 = *(const float4*)(p + 4);
  const float4 x2 = *(const float4*)(q);
  const float4 x3 = *(const float4*)(q + 4);
  v16bf a;
  a[0]  = f2bf(x0.x); a[1]  = f2bf(x0.y); a[2]  = f2bf(x0.z); a[3]  = f2bf(x0.w);
  a[4]  = f2bf(x1.x); a[5]  = f2bf(x1.y); a[6]  = f2bf(x1.z); a[7]  = f2bf(x1.w);
  a[8]  = f2bf(x2.x); a[9]  = f2bf(x2.y); a[10] = f2bf(x2.z); a[11] = f2bf(x2.w);
  a[12] = f2bf(x3.x); a[13] = f2bf(x3.y); a[14] = f2bf(x3.z); a[15] = f2bf(x3.w);
  return a;
}

// ---------------------------------------------------------------------------
// Batched WMMA GEMM:  C = alpha*(A @ B) [+ bias] [gelu] [+ resid]
//   A: [M,Kc] row-major (lda); B: [Kc,Nc] (ldb) or bT=1 -> B[k,n]=Bm[n*ldb+k]
//   batch z (gridDim.y): A += z*sA, B += z*sB, C += z*sC, resid += z*sR
//   8 waves / 256-thread block; each wave owns a 32x64 output macro-tile:
//   2 A frags x 4 B frags -> 8 v_wmma_f32_16x16x32_bf16 per 32-deep K step.
//   Fragments are built straight from global memory (no LDS, no barriers).
// ---------------------------------------------------------------------------
__global__ __launch_bounds__(256) void gemm_wmma_kernel(
    const float* __restrict__ A, int lda, long sA,
    const float* __restrict__ Bm, int ldb, long sB, int bT,
    const float* __restrict__ bias,
    const float* __restrict__ resid, long sR,
    float* __restrict__ C, int ldc, long sC,
    int M, int Nc, int Kc, float alpha, int act)
{
  const int z = blockIdx.y;
  A  += (long)z * sA;
  Bm += (long)z * sB;
  C  += (long)z * sC;
  if (resid) resid += (long)z * sR;

  const int ntN = (Nc + 63) >> 6;
  const int ntM = (M + 31) >> 5;
  const int wave = threadIdx.x >> 5;
  const int lane = threadIdx.x & 31;
  const int tile = blockIdx.x * 8 + wave;
  if (tile >= ntM * ntN) return;

  const int tm = tile / ntN;
  const int tn = tile % ntN;
  const int m0 = tm * 32, n0 = tn * 64;
  const int half = lane >> 4;
  const int lmod = lane & 15;

  const bool kfull = (Kc & 31) == 0;
  const bool fastA = kfull && ((lda & 3) == 0) && (m0 + 32 <= M);
  const bool fastB = kfull && bT && ((ldb & 3) == 0) && (n0 + 64 <= Nc);

  v8f zf = {0.f, 0.f, 0.f, 0.f, 0.f, 0.f, 0.f, 0.f};
  v8f acc[8];
  #pragma unroll
  for (int i = 0; i < 8; ++i) acc[i] = zf;

  const int aoff = half ? 8 : 0;    // K sub-offset for A fragment
  const int boff = half ? 8 : 0;    // K sub-offset for B fragment
  const int ma = m0 + lmod;
  const int mb = m0 + 16 + lmod;

  for (int k0 = 0; k0 < Kc; k0 += 32) {
    v16bf a0, a1, bf[4];

    if (fastA) {
      const float* p0 = A + (long)ma * lda + k0 + aoff;
      const float* p1 = A + (long)mb * lda + k0 + aoff;
      a0 = cvt16(p0, p0 + 16);
      a1 = cvt16(p1, p1 + 16);
      if (k0 + 32 < Kc) __builtin_prefetch(p0 + 32, 0, 1);
    } else {
      #pragma unroll
      for (int j = 0; j < 16; ++j) {
        int kk = k0 + aoff + ((j < 8) ? j : (8 + j));   // runs {0..7},{16..23}
        float v0 = (ma < M && kk < Kc) ? A[(long)ma * lda + kk] : 0.f;
        float v1 = (mb < M && kk < Kc) ? A[(long)mb * lda + kk] : 0.f;
        a0[j] = f2bf(v0);
        a1[j] = f2bf(v1);
      }
    }

    #pragma unroll
    for (int u = 0; u < 4; ++u) {
      int nn = n0 + u * 16 + lmod;
      if (fastB) {
        const float* pb = Bm + (long)nn * ldb + k0 + boff;
        bf[u] = cvt16(pb, pb + 16);
      } else {
        #pragma unroll
        for (int j = 0; j < 16; ++j) {
          int kk = k0 + boff + ((j < 8) ? j : (8 + j));
          float v = 0.f;
          if (kk < Kc && nn < Nc)
            v = bT ? Bm[(long)nn * ldb + kk] : Bm[(long)kk * ldb + nn];
          bf[u][j] = f2bf(v);
        }
      }
    }

    #pragma unroll
    for (int u = 0; u < 4; ++u) {
      acc[u]     = __builtin_amdgcn_wmma_f32_16x16x32_bf16(
          false, a0, false, bf[u], (short)0, acc[u], false, false);
      acc[4 + u] = __builtin_amdgcn_wmma_f32_16x16x32_bf16(
          false, a1, false, bf[u], (short)0, acc[4 + u], false, false);
    }
  }

  // epilogue: alpha, bias, gelu, residual
  #pragma unroll
  for (int u = 0; u < 4; ++u) {
    int nn = n0 + u * 16 + lmod;
    if (nn >= Nc) continue;
    float bv = bias ? bias[nn] : 0.f;
    #pragma unroll
    for (int r = 0; r < 8; ++r) {
      int mm = m0 + r + half * 8;
      if (mm < M) {
        float v = acc[u][r] * alpha + bv;
        if (act == 1) v = gelu_f(v);
        if (resid) v += resid[(long)mm * ldc + nn];
        C[(long)mm * ldc + nn] = v;
      }
      int mm2 = m0 + 16 + r + half * 8;
      if (mm2 < M) {
        float v = acc[4 + u][r] * alpha + bv;
        if (act == 1) v = gelu_f(v);
        if (resid) v += resid[(long)mm2 * ldc + nn];
        C[(long)mm2 * ldc + nn] = v;
      }
    }
  }
}

// ---------------------------------------------------------------------------
// LayerNorm over last dim C (optionally fused residual add), one block/row
// ---------------------------------------------------------------------------
__global__ __launch_bounds__(256) void ln_kernel(
    const float* __restrict__ x, const float* __restrict__ add,
    const float* __restrict__ g, const float* __restrict__ bta,
    float* __restrict__ out, int C)
{
  long row = blockIdx.x;
  const float* xr = x + row * C;
  const float* ar = add ? add + row * C : nullptr;
  __shared__ float red[256];
  int tid = threadIdx.x;

  float s = 0.f;
  for (int c = tid; c < C; c += 256) { float v = xr[c]; if (ar) v += ar[c]; s += v; }
  red[tid] = s; __syncthreads();
  for (int o = 128; o > 0; o >>= 1) { if (tid < o) red[tid] += red[tid + o]; __syncthreads(); }
  float mean = red[0] / (float)C;
  __syncthreads();

  float v2 = 0.f;
  for (int c = tid; c < C; c += 256) {
    float v = xr[c]; if (ar) v += ar[c]; v -= mean; v2 += v * v;
  }
  red[tid] = v2; __syncthreads();
  for (int o = 128; o > 0; o >>= 1) { if (tid < o) red[tid] += red[tid + o]; __syncthreads(); }
  float rstd = rsqrtf(red[0] / (float)C + 1e-5f);

  for (int c = tid; c < C; c += 256) {
    float v = xr[c]; if (ar) v += ar[c];
    out[row * C + c] = (v - mean) * rstd * g[c] + bta[c];
  }
}

// ---------------------------------------------------------------------------
// In-place softmax over last dim C (C <= 1024), one block/row.
// The row is staged once into LDS — via async global->LDS (ASYNCcnt /
// s_wait_asynccnt) when the toolchain exposes it — so the max and exp-sum
// passes read LDS; the rescale pass re-reads the exp values from global.
// ---------------------------------------------------------------------------
__global__ __launch_bounds__(256) void softmax_kernel(float* __restrict__ x, int C)
{
  long row = blockIdx.x;
  float* xr = x + row * C;
  __shared__ float red[256];
  __shared__ float rowbuf[1024];
  int tid = threadIdx.x;

  const bool use_lds = (C <= 1024) && ((C & 3) == 0);
  if (use_lds) {
#if HAVE_ASYNC_LDS
    for (int c = tid * 4; c < C; c += 256 * 4) {
      __builtin_amdgcn_global_load_async_to_lds_b128(
          (gptr_v4i)(xr + c), (lptr_v4i)(rowbuf + c), 0, 0);
    }
    __builtin_amdgcn_s_wait_asynccnt(0);
    __syncthreads();
#else
    for (int c = tid; c < C; c += 256) rowbuf[c] = xr[c];
    __syncthreads();
#endif
  }
  const float* src = use_lds ? rowbuf : xr;

  float mx = -3.4028235e38f;
  for (int c = tid; c < C; c += 256) mx = fmaxf(mx, src[c]);
  red[tid] = mx; __syncthreads();
  for (int o = 128; o > 0; o >>= 1) { if (tid < o) red[tid] = fmaxf(red[tid], red[tid + o]); __syncthreads(); }
  mx = red[0];
  __syncthreads();

  float s = 0.f;
  for (int c = tid; c < C; c += 256) { float e = __expf(src[c] - mx); xr[c] = e; s += e; }
  red[tid] = s; __syncthreads();
  for (int o = 128; o > 0; o >>= 1) { if (tid < o) red[tid] += red[tid + o]; __syncthreads(); }
  float inv = 1.f / red[0];
  for (int c = tid; c < C; c += 256) xr[c] *= inv;
}

// ---------------------------------------------------------------------------
// KNN (top-17 smallest L2 within batch), one thread per token
// ---------------------------------------------------------------------------
__global__ __launch_bounds__(64) void knn_kernel(
    const float* __restrict__ pts, int D, int* __restrict__ outidx)
{
  int t = blockIdx.x * 64 + threadIdx.x;
  if (t >= T_TOK) return;
  int b = t >> 10;
  const float* base = pts + (long)(b << 10) * D;
  float q[32];
  for (int d = 0; d < D; ++d) q[d] = pts[(long)t * D + d];

  float bd[KNN_K]; int bi[KNN_K];
  for (int i = 0; i < KNN_K; ++i) { bd[i] = 3.4028235e38f; bi[i] = 0; }

  for (int j = 0; j < NPTS; ++j) {
    const float* pj = base + (long)j * D;
    float d2 = 0.f;
    for (int d = 0; d < D; ++d) { float df = q[d] - pj[d]; d2 += df * df; }
    if (d2 < bd[KNN_K - 1]) {
      int pos = KNN_K - 1;
      while (pos > 0 && bd[pos - 1] > d2) {
        bd[pos] = bd[pos - 1]; bi[pos] = bi[pos - 1]; --pos;
      }
      bd[pos] = d2; bi[pos] = j;
    }
  }
  for (int i = 0; i < KNN_K; ++i) outidx[(long)t * KNN_K + i] = bi[i];
}

// ---------------------------------------------------------------------------
// Fused local features: geo(6->64)+gelu+max  and  feat(64->64)+gelu+max
// one block per token, 64 threads = 64 output channels; y: [T,128]
// ---------------------------------------------------------------------------
__global__ __launch_bounds__(64) void localfeat_kernel(
    const float* __restrict__ xyz, const float* __restrict__ f32,
    const int* __restrict__ idxg, const int* __restrict__ idxf,
    const float* __restrict__ gw, const float* __restrict__ gb,
    const float* __restrict__ fw, const float* __restrict__ fb,
    float* __restrict__ y)
{
  int t = blockIdx.x;
  int c = threadIdx.x;
  long bas = (long)((t >> 10) << 10);

  __shared__ float cx[3];
  __shared__ float cf[32];
  __shared__ float nf[32];
  if (c < 3)  cx[c] = xyz[(long)t * 3 + c];
  if (c < 32) cf[c] = f32[(long)t * 32 + c];
  __syncthreads();

  // ---- geo branch ----
  float w0 = gw[c*6+0], w1 = gw[c*6+1], w2 = gw[c*6+2];
  float w3 = gw[c*6+3], w4 = gw[c*6+4], w5 = gw[c*6+5];
  float gbias = gb[c];
  float gmax = -3.4028235e38f;
  for (int k = 0; k < KNN_K; ++k) {
    int j = idxg[(long)t * KNN_K + k];
    float nx = xyz[(bas + j) * 3 + 0];
    float ny = xyz[(bas + j) * 3 + 1];
    float nz = xyz[(bas + j) * 3 + 2];
    float a = w0*cx[0] + w1*cx[1] + w2*cx[2]
            + w3*(nx - cx[0]) + w4*(ny - cx[1]) + w5*(nz - cx[2]) + gbias;
    gmax = fmaxf(gmax, gelu_f(a));
  }

  // ---- feature branch ----
  float a0 = fb[c];
  for (int i = 0; i < 32; ++i) a0 += fw[c*64 + i] * cf[i];
  float fmx = -3.4028235e38f;
  for (int k = 0; k < KNN_K; ++k) {
    int j = idxf[(long)t * KNN_K + k];
    if (c < 32) nf[c] = f32[(bas + j) * 32 + c];
    __syncthreads();
    float a = a0;
    for (int i = 0; i < 32; ++i) a += fw[c*64 + 32 + i] * (nf[i] - cf[i]);
    fmx = fmaxf(fmx, gelu_f(a));
    __syncthreads();
  }

  y[(long)t * 128 + c]      = gmax;
  y[(long)t * 128 + 64 + c] = fmx;
}

// ---------------------------------------------------------------------------
// Small data-movement kernels
// ---------------------------------------------------------------------------
__global__ void tr_cn_kernel(const float* __restrict__ src, float* __restrict__ dst)
{ // (B,3,N) -> [T,3]
  int i = blockIdx.x * 256 + threadIdx.x;
  if (i >= T_TOK * 3) return;
  int t = i / 3, c = i % 3;
  int b = t >> 10, n = t & 1023;
  dst[i] = src[((long)b * 3 + c) * NPTS + n];
}

__global__ void concat_y0_kernel(const float* __restrict__ y,
                                 const float* __restrict__ fg,
                                 float* __restrict__ y0)
{ // y0[t,0:128]=y, y0[t,128:256]=fg[batch(t)]
  long i = (long)blockIdx.x * 256 + threadIdx.x;
  if (i >= (long)T_TOK * 256) return;
  long t = i >> 8; int c = (int)(i & 255);
  y0[i] = (c < 128) ? y[t * 128 + c] : fg[(long)(t >> 10) * 128 + (c - 128)];
}

__global__ void pack_ps_kernel(const float* __restrict__ ps, float* __restrict__ psp)
{ // ps_w (512,128,4) -> 4 matrices [512,128]
  int i = blockIdx.x * 256 + threadIdx.x;
  if (i >= 4 * 512 * 128) return;
  int r = i / (512 * 128); int rem = i % (512 * 128);
  int c = rem / 128; int o = rem % 128;
  psp[i] = ps[(long)c * 512 + o * 4 + r];
}

__global__ void concat_delta_kernel(const float* __restrict__ y3up,
                                    const float* __restrict__ y,
                                    float* __restrict__ cat4)
{
  long i = (long)blockIdx.x * 256 + threadIdx.x;
  if (i >= (long)T4_TOK * 256) return;
  long t4 = i >> 8; int c = (int)(i & 255);
  cat4[i] = (c < 128) ? y3up[t4 * 128 + c] : y[(t4 >> 2) * 128 + (c - 128)];
}

__global__ void write_out_kernel(const float* __restrict__ res3,
                                 const float* __restrict__ coarse,
                                 float* __restrict__ dout)
{ // out (B,3,4096) = res3 + repeat(coarse,4)
  int i = blockIdx.x * 256 + threadIdx.x;
  if (i >= BATCH * 3 * 4096) return;
  int b = i / (3 * 4096); int rem = i % (3 * 4096);
  int c = rem / 4096; int m = rem % 4096;
  long t4 = (long)b * 4096 + m;
  dout[i] = res3[t4 * 3 + c] + coarse[((long)b * 3 + c) * NPTS + (m >> 2)];
}

__global__ void write_y3_kernel(const float* __restrict__ y3up, float* __restrict__ dout)
{ // y3 (B,128,4096) from token-major [T4,128]
  long i = (long)blockIdx.x * 256 + threadIdx.x;
  if (i >= (long)BATCH * 128 * 4096) return;
  long b = i / (128 * 4096); long rem = i % (128 * 4096);
  long c = rem / 4096; long m = rem % 4096;
  dout[i] = y3up[((long)b * 4096 + m) * 128 + c];
}

// ---------------------------------------------------------------------------
// Host-side helpers
// ---------------------------------------------------------------------------
static inline void gemm(hipStream_t st,
                        const float* A, int lda, long sA,
                        const float* B, int ldb, long sB, int bT,
                        const float* bias, const float* resid, long sR,
                        float* C, int ldc, long sC,
                        int M, int Nc, int Kc, float alpha, int act, int nb)
{
  int nt = ((M + 31) / 32) * ((Nc + 63) / 64);
  dim3 grid((nt + 7) / 8, nb);
  gemm_wmma_kernel<<<grid, 256, 0, st>>>(A, lda, sA, B, ldb, sB, bT,
                                         bias, resid, sR, C, ldc, sC,
                                         M, Nc, Kc, alpha, act);
}

// transformer block, params at d_in[pb .. pb+13]
static inline void run_block(hipStream_t st, void* const* d_in, int pb,
                             const float* in, int din, float* out,
                             float* sbuf, float* t0, float* qkv, float* scores,
                             float* obuf, float* attn, float* t1, float* h1)
{
  const float* proj_w = (const float*)d_in[pb + 0];
  const float* proj_b = (const float*)d_in[pb + 1];
  const float* ln13_g = (const float*)d_in[pb + 2];
  const float* ln13_b = (const float*)d_in[pb + 3];
  const float* in_w   = (const float*)d_in[pb + 4];
  const float* in_b   = (const float*)d_in[pb + 5];
  const float* out_w  = (const float*)d_in[pb + 6];
  const float* out_b  = (const float*)d_in[pb + 7];
  const float* ln12_g = (const float*)d_in[pb + 8];
  const float* ln12_b = (const float*)d_in[pb + 9];
  const float* l11_w  = (const float*)d_in[pb + 10];
  const float* l11_b  = (const float*)d_in[pb + 11];
  const float* l12_w  = (const float*)d_in[pb + 12];
  const float* l12_b  = (const float*)d_in[pb + 13];

  // s = in @ proj_w^T + proj_b      [T,512]
  gemm(st, in, din, 0, proj_w, din, 0, 1, proj_b, nullptr, 0,
       sbuf, 512, 0, T_TOK, 512, din, 1.f, 0, 1);
  // t0 = LN(s)
  ln_kernel<<<T_TOK, 256, 0, st>>>(sbuf, nullptr, ln13_g, ln13_b, t0, 512);
  // qkv = t0 @ in_w^T + in_b        [T,1536]
  gemm(st, t0, 512, 0, in_w, 512, 0, 1, in_b, nullptr, 0,
       qkv, 1536, 0, T_TOK, 1536, 512, 1.f, 0, 1);

  // attention per batch; heads batched via gridDim.y (stride 128 inside qkv)
  const float scale = 0.08838834764831845f; // 1/sqrt(128)
  for (int b = 0; b < BATCH; ++b) {
    const float* qb = qkv + (long)b * NPTS * 1536;
    // scores[h] = (Q_h @ K_h^T) * scale   [4,1024,1024]
    gemm(st, qb, 1536, 128, qb + 512, 1536, 128, 1, nullptr, nullptr, 0,
         scores, NPTS, (long)NPTS * NPTS, NPTS, NPTS, 128, scale, 0, 4);
    softmax_kernel<<<4 * NPTS, 256, 0, st>>>(scores, NPTS);
    // o[b, :, h*128:...] = P_h @ V_h
    gemm(st, scores, NPTS, (long)NPTS * NPTS, qb + 1024, 1536, 128, 0,
         nullptr, nullptr, 0, obuf + (long)b * NPTS * 512, 512, 128,
         NPTS, 128, NPTS, 1.f, 0, 4);
  }

  // attn = o @ out_w^T + out_b
  gemm(st, obuf, 512, 0, out_w, 512, 0, 1, out_b, nullptr, 0,
       attn, 512, 0, T_TOK, 512, 512, 1.f, 0, 1);
  // t1 = LN(t0 + attn)
  ln_kernel<<<T_TOK, 256, 0, st>>>(t0, attn, ln12_g, ln12_b, t1, 512);
  // h1 = gelu(t1 @ l11^T + b)
  gemm(st, t1, 512, 0, l11_w, 512, 0, 1, l11_b, nullptr, 0,
       h1, 1024, 0, T_TOK, 1024, 512, 1.f, 1, 1);
  // out = t1 + (h1 @ l12^T + b)
  gemm(st, h1, 1024, 0, l12_w, 1024, 0, 1, l12_b, t1, 0,
       out, 512, 0, T_TOK, 512, 1024, 1.f, 0, 1);
}

// ---------------------------------------------------------------------------
// kernel_launch
// ---------------------------------------------------------------------------
extern "C" void kernel_launch(void* const* d_in, const int* in_sizes, int n_in,
                              void* d_out, int out_size, void* d_ws, size_t ws_size,
                              hipStream_t stream)
{
  (void)in_sizes; (void)n_in; (void)out_size; (void)ws_size;

  const float* coarse = (const float*)d_in[1];
  const float* featg  = (const float*)d_in[2];   // (8,512,1) == [8,512]
  const float* mlp_w  = (const float*)d_in[3];
  const float* mlp_b  = (const float*)d_in[4];
  const float* geo_w  = (const float*)d_in[5];
  const float* geo_b  = (const float*)d_in[6];
  const float* feat_w = (const float*)d_in[7];
  const float* feat_b = (const float*)d_in[8];
  const float* c11_w  = (const float*)d_in[9];
  const float* c11_b  = (const float*)d_in[10];
  const float* c1_w   = (const float*)d_in[11];
  const float* c1_b   = (const float*)d_in[12];
  const int P_SA1 = 13, P_SA2 = 27, P_SA3 = 41;
  const float* ps_w    = (const float*)d_in[55];
  const float* delta_w = (const float*)d_in[56];
  const float* delta_b = (const float*)d_in[57];
  const float* o1_w    = (const float*)d_in[58];
  const float* o1_b    = (const float*)d_in[59];
  const float* o_w     = (const float*)d_in[60];
  const float* o_b     = (const float*)d_in[61];

  float* dout = (float*)d_out;

  // ---- workspace bump allocator (floats) ----
  float* W = (float*)d_ws;
  size_t off = 0;
  auto alloc = [&](size_t n) { float* p = W + off; off += n; return p; };

  float* xyz    = alloc((size_t)T_TOK * 3);
  float* feat32 = alloc((size_t)T_TOK * 32);
  int*   idxg   = (int*)alloc((size_t)T_TOK * KNN_K);
  int*   idxf   = (int*)alloc((size_t)T_TOK * KNN_K);
  float* y      = alloc((size_t)T_TOK * 128);
  float* g1     = alloc((size_t)BATCH * 256);
  float* fg     = alloc((size_t)BATCH * 128);
  float* y0     = alloc((size_t)T_TOK * 256);
  float* blkA   = alloc((size_t)T_TOK * 512);
  float* blkB   = alloc((size_t)T_TOK * 512);
  float* sbuf   = alloc((size_t)T_TOK * 512);
  float* t0     = alloc((size_t)T_TOK * 512);
  float* qkv    = alloc((size_t)T_TOK * 1536);
  float* scores = alloc((size_t)4 * NPTS * NPTS);
  float* obuf   = alloc((size_t)T_TOK * 512);
  float* attn   = alloc((size_t)T_TOK * 512);
  float* t1     = alloc((size_t)T_TOK * 512);
  float* h1     = alloc((size_t)T_TOK * 1024);
  float* psp    = alloc((size_t)4 * 512 * 128);
  float* y3up   = alloc((size_t)T4_TOK * 128);
  float* cat4   = alloc((size_t)T4_TOK * 256);
  float* y4     = alloc((size_t)T4_TOK * 128);
  float* h64    = alloc((size_t)T4_TOK * 64);
  float* res3   = alloc((size_t)T4_TOK * 3);

  // ---- front-end ----
  tr_cn_kernel<<<(T_TOK * 3 + 255) / 256, 256, 0, stream>>>(coarse, xyz);
  // feature = gelu(xyz @ mlp_w^T + mlp_b)   [T,32]
  gemm(stream, xyz, 3, 0, mlp_w, 3, 0, 1, mlp_b, nullptr, 0,
       feat32, 32, 0, T_TOK, 32, 3, 1.f, 1, 1);
  knn_kernel<<<T_TOK / 64, 64, 0, stream>>>(xyz, 3, idxg);
  knn_kernel<<<T_TOK / 64, 64, 0, stream>>>(feat32, 32, idxf);
  localfeat_kernel<<<T_TOK, 64, 0, stream>>>(xyz, feat32, idxg, idxf,
                                             geo_w, geo_b, feat_w, feat_b, y);

  // global feature: fg = (gelu(featg @ c11^T + b)) @ c1^T + b   [8,128]
  gemm(stream, featg, 512, 0, c11_w, 512, 0, 1, c11_b, nullptr, 0,
       g1, 256, 0, BATCH, 256, 512, 1.f, 1, 1);
  gemm(stream, g1, 256, 0, c1_w, 256, 0, 1, c1_b, nullptr, 0,
       fg, 128, 0, BATCH, 128, 256, 1.f, 0, 1);

  concat_y0_kernel<<<(int)(((size_t)T_TOK * 256 + 255) / 256), 256, 0, stream>>>(y, fg, y0);

  // ---- transformer stack ----
  run_block(stream, d_in, P_SA1, y0,   256, blkA, sbuf, t0, qkv, scores, obuf, attn, t1, h1);
  run_block(stream, d_in, P_SA2, blkA, 512, blkB, sbuf, t0, qkv, scores, obuf, attn, t1, h1);
  run_block(stream, d_in, P_SA3, blkB, 512, blkA, sbuf, t0, qkv, scores, obuf, attn, t1, h1);
  float* y3t = blkA;   // [T,512] token-major

  // ---- ConvTranspose1d(512,128,4,4) as 4 GEMMs ----
  pack_ps_kernel<<<(4 * 512 * 128 + 255) / 256, 256, 0, stream>>>(ps_w, psp);
  for (int r = 0; r < 4; ++r) {
    gemm(stream, y3t, 512, 0, psp + (size_t)r * 512 * 128, 128, 0, 0,
         nullptr, nullptr, 0, y3up + r * 128, 512, 0,
         T_TOK, 128, 512, 1.f, 0, 1);
  }

  // ---- delta / output head ----
  concat_delta_kernel<<<(int)(((size_t)T4_TOK * 256 + 255) / 256), 256, 0, stream>>>(y3up, y, cat4);
  gemm(stream, cat4, 256, 0, delta_w, 256, 0, 1, delta_b, nullptr, 0,
       y4, 128, 0, T4_TOK, 128, 256, 1.f, 0, 1);
  gemm(stream, y4, 128, 0, o1_w, 128, 0, 1, o1_b, nullptr, 0,
       h64, 64, 0, T4_TOK, 64, 128, 1.f, 1, 1);
  gemm(stream, h64, 64, 0, o_w, 64, 0, 1, o_b, nullptr, 0,
       res3, 3, 0, T4_TOK, 3, 64, 1.f, 0, 1);

  // ---- write outputs: out (B,3,4096) then y3 (B,128,4096) ----
  write_out_kernel<<<(BATCH * 3 * 4096 + 255) / 256, 256, 0, stream>>>(res3, coarse, dout);
  write_y3_kernel<<<(int)(((size_t)BATCH * 128 * 4096 + 255) / 256), 256, 0, stream>>>(
      y3up, dout + (size_t)BATCH * 3 * 4096);
}
